// LSTM_3281355014150
// MI455X (gfx1250) — matrix operand order repaired and checked by reading, not compiled, not run on hardware
//
#include <hip/hip_runtime.h>

// ---------------------------------------------------------------------------
// LSTM cell, N=4096, F=2048, B=2048.  Strategy:
//   1) convert h0|behavior -> bf16 A[4096][4096]          (ws)
//   2) transpose-convert gate weights -> bf16 Wt[g][n][k] (ws, K-major)
//   3) fused 4-gate GEMM (v_wmma_f32_16x16x32_bf16) + LSTM epilogue,
//      double-buffered LDS fed by GLOBAL_LOAD_ASYNC_TO_LDS_B128 (ASYNCcnt)
// bf16 working set ~101MB -> resident in 192MB L2; compute-bound on WMMA.
// ---------------------------------------------------------------------------

typedef __bf16 bf16_t;
typedef __attribute__((ext_vector_type(16))) __bf16 v16bf;
typedef __attribute__((ext_vector_type(8)))  float  v8f;
typedef __attribute__((ext_vector_type(4)))  float  f32x4;

union Frag { v16bf v; f32x4 f[2]; };

#define NROWS 4096   // batch N
#define FDIM  2048   // feature / output cols per gate
#define KTOT  4096   // fused K = F (h) + B (x)
#define BM    128
#define BN    64
#define KT    64

__device__ __forceinline__ float sigmoid_fast(float x) {
    return 1.0f / (1.0f + __expf(-x));
}
__device__ __forceinline__ float tanh_fast(float x) {
    float ax = fabsf(x);
    float e  = __expf(-2.0f * ax);
    float t  = (1.0f - e) / (1.0f + e);
    return copysignf(t, x);
}

// Async global->LDS copy (CDNA5 async engine, tracked by ASYNCcnt).
// GVS addressing: mem = SADDR(64) + VADDR(i32) + offset ; lds = VDST(u32).
__device__ __forceinline__ void async_ld_b128(unsigned lds, int voff,
                                              const void* base) {
    asm volatile("global_load_async_to_lds_b128 %0, %1, %2 offset:0"
                 :: "v"(lds), "v"(voff), "s"(base) : "memory");
}
__device__ __forceinline__ void wait_async0() {
    asm volatile("s_wait_asynccnt 0" ::: "memory");
}

// ---- 1) activations fp32 -> bf16, concatenated [h0 | behavior] ------------
__global__ __launch_bounds__(256) void convert_act(const float* __restrict__ h0,
                                                   const float* __restrict__ xb,
                                                   bf16_t* __restrict__ A) {
    size_t t   = (size_t)blockIdx.x * 256 + threadIdx.x;
    size_t idx = t * 4;                      // 4 elements / thread
    int row = (int)(idx >> 12);              // /4096
    int col = (int)(idx & 4095);
    const float* src = (col < FDIM) ? (h0 + (size_t)row * FDIM + col)
                                    : (xb + (size_t)row * FDIM + (col - FDIM));
    f32x4 v = *(const f32x4*)src;
    A[idx + 0] = (bf16_t)v.x;
    A[idx + 1] = (bf16_t)v.y;
    A[idx + 2] = (bf16_t)v.z;
    A[idx + 3] = (bf16_t)v.w;
}

// ---- 2) weight fp32 [k][n] -> bf16 Wt[n][KTOT] at column offset koff ------
__global__ __launch_bounds__(256) void convert_wt(const float* __restrict__ W,
                                                  bf16_t* __restrict__ Wt,
                                                  int koff) {
    __shared__ float tile[32][33];
    int k0 = blockIdx.x * 32, n0 = blockIdx.y * 32;
    int tx = threadIdx.x, ty = threadIdx.y;          // 32 x 8
#pragma unroll
    for (int i = 0; i < 4; ++i)
        tile[ty + 8 * i][tx] = W[(size_t)(k0 + ty + 8 * i) * FDIM + n0 + tx];
    __syncthreads();
#pragma unroll
    for (int i = 0; i < 4; ++i)
        Wt[(size_t)(n0 + ty + 8 * i) * KTOT + koff + k0 + tx] =
            (bf16_t)tile[tx][ty + 8 * i];
}

// ---- 3) fused 4-gate GEMM + LSTM epilogue ---------------------------------
__global__ __launch_bounds__(256) void lstm_gemm(
    const bf16_t* __restrict__ A,    // [4096][4096]
    const bf16_t* __restrict__ Wt,   // 4 x [2048][4096]
    const float*  __restrict__ c0,
    const float*  __restrict__ bi, const float* __restrict__ bf_,
    const float*  __restrict__ bg, const float* __restrict__ bo,
    float* __restrict__ out)
{
    __shared__ __align__(16) bf16_t As[2][BM * KT];          // 32 KB
    __shared__ __align__(16) bf16_t Bs[2][4][BN * KT];       // 64 KB

    const int tid  = threadIdx.x;
    const int lane = tid & 31;
    const int wave = tid >> 5;
    const int wm   = wave >> 1;          // 0..3  (M direction)
    const int wn   = wave & 1;           // 0..1  (N direction)
    const int bm   = blockIdx.x * BM;
    const int bn   = blockIdx.y * BN;

    v8f acc[4][2][2] = {};               // [gate][mt][nt]

    // Per-thread async-copy descriptors: 4 A chunks + 8 B chunks of 16B.
    int      gA[4], gB[8];               // global byte offsets (advance 128/iter)
    unsigned lA[4], lB[8];               // LDS byte addresses (buffer 0)
#pragma unroll
    for (int i = 0; i < 4; ++i) {
        int c = tid + i * 256;
        int row = c >> 3, kc = (c & 7) * 8;
        gA[i] = (int)(((size_t)(bm + row) * KTOT + kc) * 2);
        lA[i] = (unsigned)(uintptr_t)&As[0][c * 8];
    }
#pragma unroll
    for (int i = 0; i < 8; ++i) {
        int c = tid + i * 256;
        int g = c >> 9, r = (c >> 3) & 63, kc = (c & 7) * 8;
        gB[i] = (int)(((size_t)g * FDIM * KTOT + (size_t)(bn + r) * KTOT + kc) * 2);
        lB[i] = (unsigned)(uintptr_t)&Bs[0][0][c * 8];
    }
    const unsigned A_BUF = BM * KT * 2;          // 16 KB per A buffer
    const unsigned B_BUF = 4 * BN * KT * 2;      // 32 KB per B buffer

    auto issue = [&](int buf) {
#pragma unroll
        for (int i = 0; i < 4; ++i) {
            async_ld_b128(lA[i] + (unsigned)buf * A_BUF, gA[i], A);
            gA[i] += KT * 2;
        }
#pragma unroll
        for (int i = 0; i < 8; ++i) {
            async_ld_b128(lB[i] + (unsigned)buf * B_BUF, gB[i], Wt);
            gB[i] += KT * 2;
        }
    };

    issue(0);
    wait_async0();
    __syncthreads();

    const int NK = KTOT / KT;            // 64
    const int rA = wm * 32 + (lane & 15);
    const int rB = wn * 32 + (lane & 15);
    const int kA = (lane >> 4) * 8;      // A: runs of 8 at kA and kA+16
    const int kB = (lane >> 4) * 8;      // B: runs of 8 at kB and kB+16

    for (int kt = 0; kt < NK; ++kt) {
        const int cur = kt & 1;
        if (kt + 1 < NK) issue(cur ^ 1); // async fill of next buffer

#pragma unroll
        for (int ks = 0; ks < 2; ++ks) {             // two 32-deep WMMA steps
            Frag a[2];
#pragma unroll
            for (int mt = 0; mt < 2; ++mt) {
                const bf16_t* p = &As[cur][(rA + mt * 16) * KT + ks * 32 + kA];
                a[mt].f[0] = *(const f32x4*)p;
                a[mt].f[1] = *(const f32x4*)(p + 16);
            }
#pragma unroll
            for (int g = 0; g < 4; ++g) {
#pragma unroll
                for (int nt = 0; nt < 2; ++nt) {
                    Frag b;
                    const bf16_t* p = &Bs[cur][g][(rB + nt * 16) * KT + ks * 32 + kB];
                    b.f[0] = *(const f32x4*)p;
                    b.f[1] = *(const f32x4*)(p + 16);
#pragma unroll
                    for (int mt = 0; mt < 2; ++mt) {
                        acc[g][mt][nt] = __builtin_amdgcn_wmma_f32_16x16x32_bf16(
                            false, a[mt].v, false, b.v,
                            (short)0, acc[g][mt][nt], false, false);
                    }
                }
            }
        }
        wait_async0();                   // next buffer landed in LDS
        __syncthreads();
    }

    // ---- LSTM epilogue:  c = f*c0 + i*g ;  h = o*tanh(c) ------------------
    const int colb = bn + wn * 32 + (lane & 15);
    const int rowb = bm + wm * 32 + ((lane >> 4) << 3);   // +8 for upper lanes
#pragma unroll
    for (int nt = 0; nt < 2; ++nt) {
        const int n   = colb + nt * 16;
        const float vbi = bi[n], vbf = bf_[n], vbg = bg[n], vbo = bo[n];
#pragma unroll
        for (int mt = 0; mt < 2; ++mt) {
#pragma unroll
            for (int r = 0; r < 8; ++r) {
                const int m = rowb + mt * 16 + r;
                const float iv = sigmoid_fast(acc[0][mt][nt][r] + vbi);
                const float fv = sigmoid_fast(acc[1][mt][nt][r] + vbf);
                const float gv = tanh_fast  (acc[2][mt][nt][r] + vbg);
                const float ov = sigmoid_fast(acc[3][mt][nt][r] + vbo);
                const float cc = fv * c0[(size_t)m * FDIM + n] + iv * gv;
                out[(size_t)m * FDIM + n] = ov * tanh_fast(cc);
            }
        }
    }
}

// ---------------------------------------------------------------------------
extern "C" void kernel_launch(void* const* d_in, const int* in_sizes, int n_in,
                              void* d_out, int out_size, void* d_ws, size_t ws_size,
                              hipStream_t stream) {
    const float* behavior = (const float*)d_in[0];
    const float* h0       = (const float*)d_in[1];
    const float* c0       = (const float*)d_in[2];
    const float* Wh[4] = {(const float*)d_in[3], (const float*)d_in[6],
                          (const float*)d_in[9], (const float*)d_in[12]};
    const float* Wx[4] = {(const float*)d_in[4], (const float*)d_in[7],
                          (const float*)d_in[10], (const float*)d_in[13]};
    const float* bia[4] = {(const float*)d_in[5], (const float*)d_in[8],
                           (const float*)d_in[11], (const float*)d_in[14]};

    bf16_t* A  = (bf16_t*)d_ws;                              // 32 MB
    bf16_t* Wt = A + (size_t)NROWS * KTOT;                   // 64 MB (4 gates)

    convert_act<<<16384, 256, 0, stream>>>(h0, behavior, A);

    for (int g = 0; g < 4; ++g) {
        bf16_t* Wtg = Wt + (size_t)g * FDIM * KTOT;
        convert_wt<<<dim3(64, 64), dim3(32, 8), 0, stream>>>(Wh[g], Wtg, 0);
        convert_wt<<<dim3(64, 64), dim3(32, 8), 0, stream>>>(Wx[g], Wtg, FDIM);
    }

    lstm_gemm<<<dim3(NROWS / BM, FDIM / BN), 256, 0, stream>>>(
        A, Wt, c0, bia[0], bia[1], bia[2], bia[3], (float*)d_out);
}
